// StudentWaveNet_69123203662603
// MI455X (gfx1250) — matrix operand order repaired and checked by reading, not compiled, not run on hardware
//
#include <hip/hip_runtime.h>
#include <hip/hip_bf16.h>
#include <math.h>

typedef _Float16 f16;
typedef __attribute__((ext_vector_type(8)))  _Float16 v8h;
typedef __attribute__((ext_vector_type(16))) _Float16 v16h;
typedef __attribute__((ext_vector_type(8)))  float    v8f;

#define NT 64                 // time steps per workgroup
#define XPAD 72               // padded channel row (64 + 8), 144 B rows (16B aligned)
#define GPAD 40               // padded gated row (32 + 8), 80 B rows (16B aligned)
#define SQRT_HALF 0.70710678118654752440f

static __device__ __forceinline__ v16h join8(v8h lo, v8h hi) {
    union { v16h v; v8h h[2]; } u;
    u.h[0] = lo; u.h[1] = hi;
    return u.v;
}

// ---------------------------------------------------------------------------
// Weight pre-pack: rearrange conv_w / out_w (f32) into per-lane WMMA A-fragment
// order (f16) so the layer kernel loads each A fragment as 2x b128.
// A 16x32 f16 order used here (B uses the same k): lane -> M = mb*16 + lane%16, half = lane/16;
//                     element e -> K = half*16 + e
// packA index: ((((layer*3+tap)*2+kh)*4+mb)*32+lane)*16+e
// ---------------------------------------------------------------------------
__global__ void pack_conv_w_kernel(const float* __restrict__ conv_w,
                                   f16* __restrict__ packA, int n)
{
    int i = blockIdx.x * 256 + threadIdx.x;
    if (i >= n) return;
    int e    = i & 15;
    int lane = (i >> 4) & 31;
    int mb   = (i >> 9) & 3;
    int kh   = (i >> 11) & 1;
    int tap  = (i >> 12) % 3;
    int layer = i / 12288;
    int v = e >> 1, pos = e & 1;
    int m    = mb * 16 + (lane & 15);
    int half = lane >> 4;
    int kk   = half * 16 + e;
    int rc   = kh * 32 + kk;
    packA[i] = (f16)conv_w[((size_t)(layer * 64 + m) * 64 + rc) * 3 + tap];
}

// packO index: ((layer*4+mb)*32+lane)*16+e ; K = kk over the 32 gated channels
__global__ void pack_out_w_kernel(const float* __restrict__ out_w,
                                  f16* __restrict__ packO, int n)
{
    int i = blockIdx.x * 256 + threadIdx.x;
    if (i >= n) return;
    int e    = i & 15;
    int lane = (i >> 4) & 31;
    int mb   = (i >> 9) & 3;
    int layer = i >> 11;
    int v = e >> 1, pos = e & 1;
    int m    = mb * 16 + (lane & 15);
    int half = lane >> 4;
    int kk   = half * 16 + e;
    packO[i] = (f16)out_w[(size_t)layer * 64 * 32 + m * 32 + kk];
}

// ---------------------------------------------------------------------------
// Per-layer fused kernel: dilated causal conv (3x 64x64 GEMM) -> gate ->
// 64x32 projection -> residual.  8 waves / block, wave32, WMMA f16->f32.
// LDS tiles stored time-major so every B fragment is 2x ds_load_b128.
// ---------------------------------------------------------------------------
__global__ __launch_bounds__(256)
void wavenet_layer_kernel(const float* __restrict__ xin, float* __restrict__ xout,
                          const f16*  __restrict__ packA, const float* __restrict__ conv_b,
                          const f16*  __restrict__ packO, const float* __restrict__ out_b,
                          int layer, int dil, int T)
{
    __shared__ alignas(16) f16   xs[3][NT][XPAD];  // 27 KB: shifted inputs, time-major
    __shared__ alignas(16) float hs[64][NT];       // 16 KB: conv pre-activation
    __shared__ alignas(16) f16   gs[NT][GPAD];     //  5 KB: gated acts, time-major

    const int tid = threadIdx.x;
    const int b   = blockIdx.y;
    const int t0  = blockIdx.x * NT;

    const float* xb = xin + (size_t)b * 64 * T;

    // ---- stage 3 dilation-shifted tiles into LDS (f16, transposed, zero pad)
    // decomposition keeps global reads coalesced (consecutive lanes -> consecutive t)
    for (int i = tid; i < 3 * 64 * NT; i += 256) {
        int tt  = i & 63;
        int c   = (i >> 6) & 63;
        int tap = i >> 12;
        int t   = t0 + tt - (2 - tap) * dil;
        float v = (t >= 0) ? xb[(size_t)c * T + t] : 0.0f;
        xs[tap][tt][c] = (f16)v;
    }
    __syncthreads();

    const int lane = tid & 31;
    const int wave = tid >> 5;
    const int mb   = wave & 3;          // output-channel 16-block
    const int nb0  = (wave >> 2) * 2;   // first of two time 16-blocks
    const int lm   = lane & 15;
    const int half = lane >> 4;

    // ---- conv A fragments from pre-packed weights: 2x global b128 each
    v16h aw[3][2];
    {
        const f16* pw = packA + ((size_t)layer * 24 + (size_t)mb) * 512 + (size_t)lane * 16;
        #pragma unroll
        for (int tap = 0; tap < 3; ++tap) {
            #pragma unroll
            for (int kh = 0; kh < 2; ++kh) {
                const f16* p = pw + ((size_t)tap * 8 + (size_t)kh * 4) * 512;
                aw[tap][kh] = join8(*(const v8h*)p, *(const v8h*)(p + 8));
            }
        }
    }

    // ---- conv GEMM: per wave 2 output tiles x (3 taps x 2 K-halves) = 12 WMMAs
    #pragma unroll
    for (int nn = 0; nn < 2; ++nn) {
        const int nb  = nb0 + nn;
        const int row = nb * 16 + lm;   // time column this lane feeds
        v8f acc = {};
        #pragma unroll
        for (int tap = 0; tap < 3; ++tap) {
            #pragma unroll
            for (int kh = 0; kh < 2; ++kh) {
                const f16* p = &xs[tap][row][kh * 32 + half * 16];
                v16h bf = join8(*(const v8h*)p, *(const v8h*)(p + 8));
                acc = __builtin_amdgcn_wmma_f32_16x16x32_f16(
                    false, aw[tap][kh], false, bf, (short)0, acc, false, false);
            }
        }
        // C/D layout: VGPR v -> M = half*8 + v, N = lane%16
        #pragma unroll
        for (int v = 0; v < 8; ++v) {
            int m = mb * 16 + half * 8 + v;
            hs[m][nb * 16 + lm] = acc[v] + conv_b[layer * 64 + m];
        }
    }
    __syncthreads();

    // ---- gating: tanh(h[0:32]) * sigmoid(h[32:64]) -> f16 (time-major)
    for (int i = tid; i < 32 * NT; i += 256) {
        int tt = i & 63, g = i >> 6;
        float a  = hs[g][tt];
        float bg = hs[g + 32][tt];
        float gv = tanhf(a) * (1.0f / (1.0f + __expf(-bg)));
        gs[tt][g] = (f16)gv;
    }
    __syncthreads();

    // ---- projection 64x32 + residual + *sqrt(1/2)
    v16h ao;
    {
        const f16* p = packO + ((size_t)layer * 4 + (size_t)mb) * 512 + (size_t)lane * 16;
        ao = join8(*(const v8h*)p, *(const v8h*)(p + 8));
    }
    float* xo = xout + (size_t)b * 64 * T;
    #pragma unroll
    for (int nn = 0; nn < 2; ++nn) {
        const int nb  = nb0 + nn;
        const int row = nb * 16 + lm;
        const f16* p = &gs[row][half * 16];
        v16h bf = join8(*(const v8h*)p, *(const v8h*)(p + 8));
        v8f acc = {};
        acc = __builtin_amdgcn_wmma_f32_16x16x32_f16(
            false, ao, false, bf, (short)0, acc, false, false);
        #pragma unroll
        for (int v = 0; v < 8; ++v) {
            int m = mb * 16 + half * 8 + v;
            int t = t0 + nb * 16 + lm;
            float res = xb[(size_t)m * T + t];
            xo[(size_t)m * T + t] =
                (acc[v] + out_b[layer * 64 + m] + res) * SQRT_HALF;
        }
    }
}

// ---------------------------------------------------------------------------
// Flow entry: x[b,c,t] = first_w[c] * z[b,t] + first_b[c]
// ---------------------------------------------------------------------------
__global__ void wavenet_first_kernel(const float* __restrict__ z, float* __restrict__ xout,
                                     const float* __restrict__ first_w,
                                     const float* __restrict__ first_b,
                                     int flow, int T)
{
    int t = blockIdx.x * blockDim.x + threadIdx.x;
    int b = blockIdx.y;
    if (t >= T) return;
    float zv = z[(size_t)b * T + t];
    #pragma unroll 8
    for (int c = 0; c < 64; ++c) {
        xout[((size_t)b * 64 + c) * T + t] =
            first_w[flow * 64 + c] * zv + first_b[flow * 64 + c];
    }
}

// ---------------------------------------------------------------------------
// Flow exit: o = last_w @ relu(x) + last_b ; mu, scale=exp ; z = z*scale+mu
// ---------------------------------------------------------------------------
__global__ void wavenet_last_kernel(const float* __restrict__ x, float* __restrict__ z,
                                    const float* __restrict__ last_w,
                                    const float* __restrict__ last_b,
                                    float* __restrict__ m_ws, float* __restrict__ s_ws,
                                    int flow, int T)
{
    int t = blockIdx.x * blockDim.x + threadIdx.x;
    int b = blockIdx.y;
    if (t >= T) return;
    float a0 = last_b[flow * 2 + 0];
    float a1 = last_b[flow * 2 + 1];
    const float* xb = x + (size_t)b * 64 * T;
    #pragma unroll 8
    for (int c = 0; c < 64; ++c) {
        float xv = fmaxf(xb[(size_t)c * T + t], 0.0f);
        a0 += last_w[(flow * 2 + 0) * 64 + c] * xv;
        a1 += last_w[(flow * 2 + 1) * 64 + c] * xv;
    }
    float scale = expf(a1);
    size_t idx = (size_t)b * T + t;
    m_ws[(size_t)flow * 8 * T + idx] = a0;
    s_ws[(size_t)flow * 8 * T + idx] = scale;
    z[idx] = z[idx] * scale + a0;
}

// ---------------------------------------------------------------------------
// Combine 4 flows: mu_tot (Horner), scale_tot (product)
// ---------------------------------------------------------------------------
__global__ void wavenet_combine_kernel(const float* __restrict__ m_ws,
                                       const float* __restrict__ s_ws,
                                       float* __restrict__ out, int T)
{
    int t = blockIdx.x * blockDim.x + threadIdx.x;
    int b = blockIdx.y;
    if (t >= T) return;
    size_t idx = (size_t)b * T + t;
    size_t N = (size_t)8 * T;
    float m0 = m_ws[idx], m1 = m_ws[N + idx], m2 = m_ws[2 * N + idx], m3 = m_ws[3 * N + idx];
    float s0 = s_ws[idx], s1 = s_ws[N + idx], s2 = s_ws[2 * N + idx], s3 = s_ws[3 * N + idx];
    float mu = ((m0 * s1 + m1) * s2 + m2) * s3 + m3;
    float st = s0 * s1 * s2 * s3;
    out[idx]     = mu;
    out[N + idx] = st;
}

// ---------------------------------------------------------------------------
extern "C" void kernel_launch(void* const* d_in, const int* in_sizes, int n_in,
                              void* d_out, int out_size, void* d_ws, size_t ws_size,
                              hipStream_t stream)
{
    const float* z_in    = (const float*)d_in[0];
    const float* conv_w  = (const float*)d_in[1];
    const float* conv_b  = (const float*)d_in[2];
    const float* out_w   = (const float*)d_in[3];
    const float* out_b   = (const float*)d_in[4];
    const float* first_w = (const float*)d_in[5];
    const float* first_b = (const float*)d_in[6];
    const float* last_w  = (const float*)d_in[7];
    const float* last_b  = (const float*)d_in[8];

    const int T = 32000, B = 8;
    const int nPackA = 60 * 3 * 2 * 4 * 32 * 16;   // 737280
    const int nPackO = 60 * 4 * 32 * 16;           // 122880

    // workspace carve-out: z | xA | xB | mu[4] | scale[4] | packA | packO (~142 MB)
    float* ws    = (float*)d_ws;
    float* z_ws  = ws;
    float* xA    = z_ws + (size_t)B * T;
    float* xB    = xA + (size_t)B * 64 * T;
    float* m_ws  = xB + (size_t)B * 64 * T;
    float* s_ws  = m_ws + (size_t)4 * B * T;
    f16*   packA = (f16*)(s_ws + (size_t)4 * B * T);
    f16*   packO = packA + nPackA;

    hipMemcpyAsync(z_ws, z_in, (size_t)B * T * sizeof(float),
                   hipMemcpyDeviceToDevice, stream);

    pack_conv_w_kernel<<<(nPackA + 255) / 256, 256, 0, stream>>>(conv_w, packA, nPackA);
    pack_out_w_kernel<<<(nPackO + 255) / 256, 256, 0, stream>>>(out_w, packO, nPackO);

    const int flow_sizes[4] = {10, 10, 10, 30};
    dim3 eb(256), eg((T + 255) / 256, B);
    dim3 lb(256), lg(T / NT, B);

    int layer = 0;
    for (int fi = 0; fi < 4; ++fi) {
        wavenet_first_kernel<<<eg, eb, 0, stream>>>(z_ws, xA, first_w, first_b, fi, T);
        float* cur = xA;
        float* nxt = xB;
        for (int l = 0; l < flow_sizes[fi]; ++l) {
            int dil = 1 << (l % 10);
            wavenet_layer_kernel<<<lg, lb, 0, stream>>>(cur, nxt, packA, conv_b,
                                                        packO, out_b, layer, dil, T);
            float* tmp = cur; cur = nxt; nxt = tmp;
            ++layer;
        }
        wavenet_last_kernel<<<eg, eb, 0, stream>>>(cur, z_ws, last_w, last_b,
                                                   m_ws, s_ws, fi, T);
    }
    wavenet_combine_kernel<<<eg, eb, 0, stream>>>(m_ws, s_ws, (float*)d_out, T);
}
